// ByteSequenceEmbedder_47450798686217
// MI455X (gfx1250) — compile-verified, matrix operand order checked by reading, not run on hardware
//
#include <hip/hip_runtime.h>

// Problem constants (match reference)
#define B_   8
#define T_   4096
#define W_   1024
#define DB_  64
#define DW_  512
#define BT_  (B_ * T_)
#define BW_  (B_ * W_)

typedef _Float16 v8h  __attribute__((ext_vector_type(8)));
typedef _Float16 v16h __attribute__((ext_vector_type(16)));
typedef float    v8f  __attribute__((ext_vector_type(8)));
typedef int      vi4  __attribute__((vector_size(4 * sizeof(int))));

#define TILE_M 128
#define TILE_N 128
#define KSTEP  32
#define LDS_PAD 8
#define LDS_STRIDE (KSTEP + LDS_PAD)   // 40 halfs = 80B rows: 16B aligned, conflict-free

enum { OP_NONE = 0, OP_RELU = 1, OP_SIGMOID = 2, OP_RESID_RELU = 3, OP_HIGHWAY = 4 };

#ifndef __has_builtin
#define __has_builtin(x) 0
#endif
#if __has_builtin(__builtin_amdgcn_global_load_async_to_lds_b128)
#define HAVE_ASYNC 1
#else
#define HAVE_ASYNC 0
#endif

// 16-byte global -> LDS copy; async (ASYNCcnt-tracked, no VGPR staging) when available.
// Probe-confirmed builtin signature: (int4* src, int4* dst, int offset, int cpol).
__device__ __forceinline__ void async16(const _Float16* g, _Float16* l) {
#if HAVE_ASYNC
  __builtin_amdgcn_global_load_async_to_lds_b128((vi4*)g, (vi4*)l, 0, 0);
#else
  *(v8h*)l = *(const v8h*)g;
#endif
}

template <int Nw>
__device__ __forceinline__ void wait_async() {
#if HAVE_ASYNC
#if __has_builtin(__builtin_amdgcn_s_wait_asynccnt)
  __builtin_amdgcn_s_wait_asynccnt(Nw);
#else
  asm volatile("s_wait_asynccnt %0" ::"n"(Nw));
#endif
#endif
}

// Build a 16-half WMMA operand fragment from an LDS row.
// Per CDNA5 ISA 7.12.2 (16-bit A 16x32): half h -> k = (h>>3)*16 + hiOff + (h&7),
// hiOff = 0 for lanes 0-15, 8 for lanes 16-31.  Two ds_load_b128 per fragment.
__device__ __forceinline__ v16h load_frag(const _Float16* rowp, int hiOff) {
  v8h lo = *(const v8h*)(rowp + hiOff);
  v8h hi = *(const v8h*)(rowp + 16 + hiOff);
  return __builtin_shufflevector(lo, hi, 0, 1, 2, 3, 4, 5, 6, 7,
                                 8, 9, 10, 11, 12, 13, 14, 15);
}

// One K-step: 2 A-frags x 4 B-frags -> 8 v_wmma_f32_16x16x32_f16 per wave.
__device__ __forceinline__ void mma_tiles(const _Float16 (*la)[LDS_STRIDE],
                                          const _Float16 (*lb)[LDS_STRIDE],
                                          v8f (&acc)[2][4], int wm, int wn, int lane) {
  const int lm = lane & 15;
  const int hiOff = (lane >> 4) << 3;
  v16h afrag[2], bfrag[4];
#pragma unroll
  for (int f = 0; f < 2; ++f)
    afrag[f] = load_frag(&la[wm * 32 + f * 16 + lm][0], hiOff);
#pragma unroll
  for (int g = 0; g < 4; ++g)
    bfrag[g] = load_frag(&lb[wn * 64 + g * 16 + lm][0], hiOff);
#pragma unroll
  for (int f = 0; f < 2; ++f)
#pragma unroll
    for (int g = 0; g < 4; ++g)
      acc[f][g] = __builtin_amdgcn_wmma_f32_16x16x32_f16(
          false, afrag[f], false, bfrag[g], (short)0, acc[f][g], false, false);
}

// Epilogue: C/D layout = VGPR r -> M = r (+8 for lanes 16-31), N = lane%16.
// Bias is one scalar load per (lane, n-frag) since all 8 acc elements share N.
__device__ __forceinline__ void epilogue_store(const v8f (&acc)[2][4], int m_base,
                                               int n_base, int wm, int wn, int lane,
                                               int N, const float* __restrict__ bias,
                                               const _Float16* __restrict__ aux,
                                               const _Float16* __restrict__ yin,
                                               _Float16* __restrict__ outH,
                                               float* __restrict__ outF, int op) {
  const int lm = lane & 15;
  const int mo = (lane >> 4) << 3;
#pragma unroll
  for (int f = 0; f < 2; ++f) {
#pragma unroll
    for (int g = 0; g < 4; ++g) {
      int n_g = n_base + wn * 64 + g * 16 + lm;
      float bv = bias ? bias[n_g] : 0.0f;
#pragma unroll
      for (int r = 0; r < 8; ++r) {
        int m_g = m_base + wm * 32 + f * 16 + mo + r;
        size_t idx = (size_t)m_g * N + n_g;
        float v = acc[f][g][r] + bv;
        if (op == OP_RELU) {
          v = v > 0.0f ? v : 0.0f;
        } else if (op == OP_SIGMOID) {
          v = 1.0f / (1.0f + __expf(-v));
        } else if (op == OP_RESID_RELU) {
          v += (float)aux[idx];
          v = v > 0.0f ? v : 0.0f;
        } else if (op == OP_HIGHWAY) {
          v = v > 0.0f ? v : 0.0f;        // h = relu(acc + bh)
          float gv = (float)aux[idx];     // gate (sigmoid already applied)
          float yv = (float)yin[idx];     // y (== A input, K==N)
          v = gv * v + (1.0f - gv) * yv;
        }
        if (outF) outF[idx] = v;
        else      outH[idx] = (_Float16)v;
      }
    }
  }
}

// Dense (taps==1) f16 WMMA GEMM: async double-buffered LDS pipeline.
//   A  : [M, K] f16      Bt : [N, K] f16 (pre-transposed weights)
__global__ __launch_bounds__(256)
void gemm_dense_kernel(const _Float16* __restrict__ A, const _Float16* __restrict__ Bt,
                       const float* __restrict__ bias, const _Float16* __restrict__ aux,
                       _Float16* __restrict__ outH, float* __restrict__ outF,
                       int N, int K, int op) {
  __shared__ _Float16 lA[2][TILE_M][LDS_STRIDE];
  __shared__ _Float16 lB[2][TILE_N][LDS_STRIDE];

  const int tid  = threadIdx.x;
  const int lane = tid & 31;
  const int wave = tid >> 5;
  const int wm = wave & 3, wn = wave >> 2;   // 4 waves x M, 2 waves x N
  const int m_base = blockIdx.x * TILE_M;
  const int n_base = blockIdx.y * TILE_N;

  // Fixed per-thread 16B chunk coords: rows row0 and row0+64, same K offset.
  const int row0 = tid >> 2;
  const int kc0  = (tid & 3) << 3;
  const _Float16* gA0 = A  + (size_t)(m_base + row0) * K + kc0;
  const _Float16* gA1 = gA0 + (size_t)64 * K;
  const _Float16* gB0 = Bt + (size_t)(n_base + row0) * K + kc0;
  const _Float16* gB1 = gB0 + (size_t)64 * K;

  v8f acc[2][4];
#pragma unroll
  for (int f = 0; f < 2; ++f)
#pragma unroll
    for (int g = 0; g < 4; ++g)
#pragma unroll
      for (int r = 0; r < 8; ++r) acc[f][g][r] = 0.0f;

  const int nk = K / KSTEP;
  auto issue = [&](int s) {   // exactly 4 async ops per thread per stage
    int buf = s & 1;
    int kt  = s * KSTEP;
    async16(gA0 + kt, &lA[buf][row0][kc0]);
    async16(gA1 + kt, &lA[buf][row0 + 64][kc0]);
    async16(gB0 + kt, &lB[buf][row0][kc0]);
    async16(gB1 + kt, &lB[buf][row0 + 64][kc0]);
  };

  issue(0);
  for (int s = 0; s < nk; ++s) {
    if (s + 1 < nk) {
      issue(s + 1);        // prefetch next stage into other buffer
      wait_async<4>();     // ASYNCcnt in-order: <=4 outstanding => stage s landed
    } else {
      wait_async<0>();
    }
    __syncthreads();
    mma_tiles(lA[s & 1], lB[s & 1], acc, wm, wn, lane);
    __syncthreads();       // all waves done reading before buffer is rewritten
  }

  epilogue_store(acc, m_base, n_base, wm, wn, lane, N, bias, aux, A, outH, outF, op);
}

// Conv-as-GEMM (taps==3, SAME padding): K-region tap -> sequence shift tap-1;
// rows shifted outside [0,T_) contribute zeros.  T_ compile-time; tiles never
// straddle a sequence boundary (TILE_M | T_), so b / t_base hoist per block.
__global__ __launch_bounds__(256)
void gemm_conv_kernel(const _Float16* __restrict__ A, const _Float16* __restrict__ Bt,
                      const float* __restrict__ bias, const _Float16* __restrict__ resid,
                      _Float16* __restrict__ outH, int N, int KA, int taps, int op) {
  __shared__ _Float16 lA[TILE_M][LDS_STRIDE];
  __shared__ _Float16 lB[TILE_N][LDS_STRIDE];

  const int tid  = threadIdx.x;
  const int lane = tid & 31;
  const int wave = tid >> 5;
  const int wm = wave & 3, wn = wave >> 2;
  const int m_base = blockIdx.x * TILE_M;
  const int n_base = blockIdx.y * TILE_N;
  const int Ktot = KA * taps;
  const int half_taps = taps >> 1;
  const int bRow = (m_base / T_) * T_;   // sequence base row
  const int tb   = m_base - bRow;        // position within sequence

  v8f acc[2][4];
#pragma unroll
  for (int f = 0; f < 2; ++f)
#pragma unroll
    for (int g = 0; g < 4; ++g)
#pragma unroll
      for (int r = 0; r < 8; ++r) acc[f][g][r] = 0.0f;

  for (int kt = 0; kt < Ktot; kt += KSTEP) {
    const int tap   = kt / KA;
    const int klb   = kt - tap * KA;
    const int shift = tap - half_taps;

    __syncthreads();
    for (int c = tid; c < (TILE_M * KSTEP / 8); c += 256) {
      int row = c >> 2;
      int kc  = (c & 3) << 3;
      int ts  = tb + row + shift;
      v8h v;
#pragma unroll
      for (int j = 0; j < 8; ++j) v[j] = (_Float16)0;
      if ((unsigned)ts < (unsigned)T_)
        v = *(const v8h*)(A + (size_t)(bRow + ts) * KA + (size_t)(klb + kc));
      *(v8h*)&lA[row][kc] = v;
    }
    for (int c = tid; c < (TILE_N * KSTEP / 8); c += 256) {
      int row = c >> 2;
      int kc  = (c & 3) << 3;
      *(v8h*)&lB[row][kc] =
          *(const v8h*)(Bt + (size_t)(n_base + row) * (size_t)Ktot + (size_t)(kt + kc));
    }
    __syncthreads();
    mma_tiles(lA, lB, acc, wm, wn, lane);
  }

  epilogue_store(acc, m_base, n_base, wm, wn, lane, N, bias, resid, nullptr,
                 outH, nullptr, op);
}

// x0[bt,c] = emb[tok[bt],c] + (bpe? emb[4,c]) + (word? emb[3,c]), f16 out.
__global__ __launch_bounds__(256)
void embed_kernel(const int* __restrict__ tok, const unsigned char* __restrict__ bpe,
                  const unsigned char* __restrict__ wrd, const float* __restrict__ emb,
                  _Float16* __restrict__ x0, int total) {
  int i = blockIdx.x * 256 + threadIdx.x;
  if (i >= total) return;
  int bt = i >> 6;  // DB_ = 64
  int c  = i & 63;
  float v = emb[(size_t)tok[bt] * DB_ + c];
  if (bpe[bt]) v += emb[4 * DB_ + c];
  if (wrd[bt]) v += emb[3 * DB_ + c];
  x0[i] = (_Float16)v;
}

// dst[N,K] f16 = transpose of src[K,N] f32 (weights; tiny, once per launch).
__global__ __launch_bounds__(256)
void transpose_to_f16(const float* __restrict__ src, _Float16* __restrict__ dst,
                      int K, int N) {
  int i = blockIdx.x * 256 + threadIdx.x;
  if (i >= K * N) return;
  int k = i / N, n = i - k * N;
  dst[(size_t)n * K + k] = (_Float16)src[i];
}

// seg_ids sorted per row -> mark each word's start byte index.
__global__ __launch_bounds__(256)
void word_start_kernel(const int* __restrict__ seg, int* __restrict__ wstart,
                       int T, int Wn, int total) {
  int i = blockIdx.x * 256 + threadIdx.x;
  if (i >= total) return;
  int t = i % T;
  if (t == 0 || seg[i] != seg[i - 1]) wstart[(i / T) * Wn + seg[i]] = t;
}

// One block per (b, word): max over the word's byte range across DW channels.
__global__ __launch_bounds__(256)
void segmax_kernel(const _Float16* __restrict__ z, const int* __restrict__ wstart,
                   _Float16* __restrict__ s, int T, int Wn, int N) {
  int bw = blockIdx.x;
  int w  = bw % Wn;
  int b  = bw / Wn;
  int t0 = wstart[bw];
  int t1 = (w + 1 < Wn) ? wstart[bw + 1] : T;
  int len = t1 - t0;
  const _Float16* p0 = z + ((size_t)b * T + t0) * (size_t)N;
  for (int c = threadIdx.x; c < N; c += 256) {
    float m = (float)p0[c];
    for (int t = 1; t < len; ++t) {
      float v = (float)p0[(size_t)t * N + c];
      m = v > m ? v : m;
    }
    s[(size_t)bw * N + c] = (_Float16)m;
  }
}

extern "C" void kernel_launch(void* const* d_in, const int* in_sizes, int n_in,
                              void* d_out, int out_size, void* d_ws, size_t ws_size,
                              hipStream_t stream) {
  (void)in_sizes; (void)n_in; (void)out_size; (void)ws_size;
  const int* tok           = (const int*)d_in[0];
  const unsigned char* bpe = (const unsigned char*)d_in[1];
  const unsigned char* wrd = (const unsigned char*)d_in[2];
  const int* seg           = (const int*)d_in[3];
  const float* emb         = (const float*)d_in[4];
  const float* c0w  = (const float*)d_in[5];  const float* c0b  = (const float*)d_in[6];
  const float* h0wg = (const float*)d_in[7];  const float* h0bg = (const float*)d_in[8];
  const float* h0wh = (const float*)d_in[9];  const float* h0bh = (const float*)d_in[10];
  const float* c1w  = (const float*)d_in[11]; const float* c1b  = (const float*)d_in[12];
  const float* h1wg = (const float*)d_in[13]; const float* h1bg = (const float*)d_in[14];
  const float* h1wh = (const float*)d_in[15]; const float* h1bh = (const float*)d_in[16];
  const float* pw   = (const float*)d_in[17]; const float* pb   = (const float*)d_in[18];

  size_t off = 0;
  auto wsa = [&](size_t bytes) -> void* {
    void* p = (char*)d_ws + off;
    off = (off + bytes + 255) & ~(size_t)255;
    return p;
  };
  _Float16* x0 = (_Float16*)wsa((size_t)BT_ * DB_ * 2);   // 4 MB
  _Float16* Y  = (_Float16*)wsa((size_t)BT_ * DW_ * 2);   // 32 MB
  _Float16* Z  = (_Float16*)wsa((size_t)BT_ * DW_ * 2);   // 32 MB
  _Float16* G  = (_Float16*)wsa((size_t)BT_ * DW_ * 2);   // 32 MB
  _Float16* H  = (_Float16*)wsa((size_t)BT_ * DW_ * 2);   // 32 MB
  int* wstart  = (int*)wsa((size_t)BW_ * 4);
  _Float16* c0wt  = (_Float16*)wsa((size_t)DW_ * (3 * DB_) * 2);
  _Float16* h0wgt = (_Float16*)wsa((size_t)2 * DW_ * DW_ * 2);
  _Float16* h0wht = (_Float16*)wsa((size_t)2 * DW_ * DW_ * 2);
  _Float16* c1wt  = (_Float16*)wsa((size_t)DW_ * (3 * DW_) * 2);
  _Float16* h1wgt = (_Float16*)wsa((size_t)2 * DW_ * DW_ * 2);
  _Float16* h1wht = (_Float16*)wsa((size_t)2 * DW_ * DW_ * 2);
  _Float16* pwt   = (_Float16*)wsa((size_t)DW_ * DW_ * 2);
  _Float16* S = Y;  // Y dead after conv1 -> reuse for segment-max output
  const size_t WW = (size_t)DW_ * DW_;

  auto tpose = [&](const float* src, _Float16* dst, int K, int N) {
    int total = K * N;
    transpose_to_f16<<<(total + 255) / 256, 256, 0, stream>>>(src, dst, K, N);
  };
  tpose(c0w, c0wt, 3 * DB_, DW_);
  for (int i = 0; i < 2; ++i) {
    tpose(h0wg + i * WW, h0wgt + i * WW, DW_, DW_);
    tpose(h0wh + i * WW, h0wht + i * WW, DW_, DW_);
    tpose(h1wg + i * WW, h1wgt + i * WW, DW_, DW_);
    tpose(h1wh + i * WW, h1wht + i * WW, DW_, DW_);
  }
  tpose(c1w, c1wt, 3 * DW_, DW_);
  tpose(pw, pwt, DW_, DW_);

  auto conv = [&](const _Float16* A, const _Float16* Bt, const float* bias,
                  const _Float16* resid, _Float16* outH, int KA, int op) {
    dim3 grid(BT_ / TILE_M, DW_ / TILE_N);
    gemm_conv_kernel<<<grid, 256, 0, stream>>>(A, Bt, bias, resid, outH, DW_, KA, 3, op);
  };
  auto dense = [&](const _Float16* A, const _Float16* Bt, const float* bias,
                   const _Float16* aux, _Float16* outH, float* outF, int M, int op) {
    dim3 grid(M / TILE_M, DW_ / TILE_N);
    gemm_dense_kernel<<<grid, 256, 0, stream>>>(A, Bt, bias, aux, outH, outF, DW_, DW_, op);
  };

  // 1) byte + marker embeddings
  embed_kernel<<<(BT_ * DB_) / 256, 256, 0, stream>>>(tok, bpe, wrd, emb, x0, BT_ * DB_);

  // 2) conv0 (SAME, k=3) + relu -> Y
  conv(x0, c0wt, c0b, nullptr, Y, DB_, OP_RELU);

  // 3) highway block 0: Y -> H -> Y   (gate GEMM, then fused highway GEMM)
  dense(Y, h0wgt,      h0bg,       nullptr, G, nullptr, BT_, OP_SIGMOID);
  dense(Y, h0wht,      h0bh,       G,       H, nullptr, BT_, OP_HIGHWAY);
  dense(H, h0wgt + WW, h0bg + DW_, nullptr, G, nullptr, BT_, OP_SIGMOID);
  dense(H, h0wht + WW, h0bh + DW_, G,       Y, nullptr, BT_, OP_HIGHWAY);

  // 4) conv1 (k=3) + residual(Y) + relu -> Z
  conv(Y, c1wt, c1b, Y, Z, DW_, OP_RESID_RELU);

  // 5) highway block 1: Z -> H -> Z
  dense(Z, h1wgt,      h1bg,       nullptr, G, nullptr, BT_, OP_SIGMOID);
  dense(Z, h1wht,      h1bh,       G,       H, nullptr, BT_, OP_HIGHWAY);
  dense(H, h1wgt + WW, h1bg + DW_, nullptr, G, nullptr, BT_, OP_SIGMOID);
  dense(H, h1wht + WW, h1bh + DW_, G,       Z, nullptr, BT_, OP_HIGHWAY);

  // 6) ragged per-word segment max -> S (f16)
  word_start_kernel<<<BT_ / 256, 256, 0, stream>>>(seg, wstart, T_, W_, BT_);
  segmax_kernel<<<BW_, 256, 0, stream>>>(Z, wstart, S, T_, W_, DW_);

  // 7) final projection -> d_out (f32)
  dense(S, pwt, pb, nullptr, nullptr, (float*)d_out, BW_, OP_NONE);
}
